// HGT_87917980549689
// MI455X (gfx1250) — compile-verified, matrix operand order checked by reading, not run on hardware
//
#include <hip/hip_runtime.h>
#include <math.h>

// ---------------------------------------------------------------- types
typedef __attribute__((ext_vector_type(16))) __bf16  v16bf;
typedef __attribute__((ext_vector_type(8)))  float   v8f;
typedef __attribute__((ext_vector_type(8)))  unsigned short ushort8;

union BFrag { v16bf bf; ushort8 u[2]; };

__device__ __forceinline__ unsigned short f2bf(float f) {
    unsigned int u = __float_as_uint(f);
    unsigned int r = u + 0x7FFFu + ((u >> 16) & 1u);   // RNE
    return (unsigned short)(r >> 16);
}
__device__ __forceinline__ float geluf(float x) {
    return 0.5f * x * (1.0f + erff(x * 0.70710678118654752f));
}
__device__ __forceinline__ void atomicMaxF(float* addr, float val) {
    unsigned int* ua = (unsigned int*)addr;
    unsigned int old = *ua;
    while (__uint_as_float(old) < val) {
        unsigned int assumed = old;
        old = atomicCAS(ua, assumed, __float_as_uint(val));
        if (old == assumed) break;
    }
}

// ---------------------------------------------------------------- GEMM
// C[M,Nc] = epilogue( A[M,K](fp32, opt. gelu) @ Bt(bf16, stored [Nc][K]) + bias )
// MODE 0: plain   MODE 1: s*o+(1-s)*Xres   MODE 2: gelu(s*o+(1-s)*Xres)
// Double-buffered LDS, fully unrolled K (template), one barrier per k-step.
#define BM 128
#define BN 128
#define BKK 32
#define LDS_S 40   // ushort stride (80B) to dodge bank conflicts, 16B aligned

template<int KVAL, int ACTA, int MODE>
__global__ __launch_bounds__(256)
void gemm_bf16_wmma(const float* __restrict__ A, int lda,
                    const unsigned short* __restrict__ Bt,
                    const float* __restrict__ bias,
                    float* __restrict__ C, int ldc,
                    const float* __restrict__ Xres, int ldres,
                    const float* __restrict__ skipPtr,
                    int M, int Nc)
{
    __shared__ unsigned short As[2][BM * LDS_S];
    __shared__ unsigned short Bs[2][BN * LDS_S];

    const int tid  = threadIdx.x;
    const int wave = tid >> 5;
    const int lane = tid & 31;
    const int l    = lane & 15;
    const int hi   = lane >> 4;
    const int m0   = blockIdx.y * BM;
    const int n0   = blockIdx.x * BN;
    const int wm   = (wave >> 2) * 64;   // 2 waves in M
    const int wn   = (wave & 3) * 32;    // 4 waves in N

    v8f zero8 = {0.f,0.f,0.f,0.f,0.f,0.f,0.f,0.f};
    v8f acc[4][2];
#pragma unroll
    for (int mi = 0; mi < 4; ++mi)
#pragma unroll
        for (int ni = 0; ni < 2; ++ni) acc[mi][ni] = zero8;

    const int lr = tid >> 1;            // row/col this thread stages
    const int lk = (tid & 1) * 16;      // 16-element K chunk
    const int grA = m0 + lr;
    const int gcB = n0 + lr;
    const bool okA = (grA < M);
    const bool okB = (gcB < Nc);
    const float* srcA = A + (size_t)grA * lda + lk;          // + kt at use
    const unsigned short* srcB = Bt + (size_t)gcB * KVAL + lk;

    auto stageA = [&](int buf, int kt) {
        unsigned short* dst = &As[buf][lr * LDS_S + lk];
        if (okA) {
            const float* s = srcA + kt;
            if (kt + BKK < KVAL) __builtin_prefetch(s + BKK, 0, 1);
#pragma unroll
            for (int j = 0; j < 16; j += 4) {
                float4 f = *(const float4*)(s + j);
                if (ACTA) {
                    f.x = geluf(f.x); f.y = geluf(f.y);
                    f.z = geluf(f.z); f.w = geluf(f.w);
                }
                dst[j + 0] = f2bf(f.x); dst[j + 1] = f2bf(f.y);
                dst[j + 2] = f2bf(f.z); dst[j + 3] = f2bf(f.w);
            }
        } else {
#pragma unroll
            for (int j = 0; j < 16; ++j) dst[j] = 0;
        }
    };
    auto stageB = [&](int buf, int kt) {
        unsigned short* dst = &Bs[buf][lr * LDS_S + lk];
        if (okB) {
            const unsigned short* s = srcB + kt;
            if (kt + BKK < KVAL) __builtin_prefetch(s + BKK, 0, 1);
            ushort8 b0 = *(const ushort8*)s;
            ushort8 b1 = *(const ushort8*)(s + 8);
            *(ushort8*)dst       = b0;
            *(ushort8*)(dst + 8) = b1;
        } else {
#pragma unroll
            for (int j = 0; j < 16; ++j) dst[j] = 0;
        }
    };

    stageA(0, 0);
    stageB(0, 0);

    constexpr int NIT = KVAL / BKK;
    int p = 0;
#pragma unroll
    for (int it = 0; it < NIT; ++it) {
        __syncthreads();
        if (it + 1 < NIT) {              // overlap: stage next tile into other buffer
            stageA(p ^ 1, (it + 1) * BKK);
            stageB(p ^ 1, (it + 1) * BKK);
        }
        // ---- build fragments per ISA 16-bit 16x32 layout & multiply ----
        BFrag afr[4], bfr[2];
#pragma unroll
        for (int mi = 0; mi < 4; ++mi) {
            const unsigned short* q = &As[p][(wm + mi * 16 + l) * LDS_S + hi * 8];
            afr[mi].u[0] = *(const ushort8*)q;        // K = hi*8 .. hi*8+7
            afr[mi].u[1] = *(const ushort8*)(q + 16); // K = 16+hi*8 ..
        }
#pragma unroll
        for (int ni = 0; ni < 2; ++ni) {
            const unsigned short* q = &Bs[p][(wn + ni * 16 + l) * LDS_S + hi * 8];
            bfr[ni].u[0] = *(const ushort8*)q;
            bfr[ni].u[1] = *(const ushort8*)(q + 16);
        }
#pragma unroll
        for (int mi = 0; mi < 4; ++mi)
#pragma unroll
            for (int ni = 0; ni < 2; ++ni)
                acc[mi][ni] = __builtin_amdgcn_wmma_f32_16x16x32_bf16(
                    false, afr[mi].bf, false, bfr[ni].bf,
                    (short)0, acc[mi][ni], false, false);
        p ^= 1;
    }

    // ---- epilogue ----
    float s = 1.0f, oms = 0.0f;
    if (MODE >= 1) {
        float sk = skipPtr[0];
        s = 1.0f / (1.0f + __expf(-sk));
        oms = 1.0f - s;
    }
#pragma unroll
    for (int mi = 0; mi < 4; ++mi)
#pragma unroll
        for (int ni = 0; ni < 2; ++ni)
#pragma unroll
            for (int r = 0; r < 8; ++r) {
                int row = m0 + wm + mi * 16 + r + 8 * hi;
                int col = n0 + wn + ni * 16 + l;
                if (row < M && col < Nc) {
                    float v = acc[mi][ni][r] + bias[col];
                    if (MODE == 1) {
                        v = s * v + oms * Xres[(size_t)row * ldres + col];
                    } else if (MODE == 2) {
                        v = s * v + oms * Xres[(size_t)row * ldres + col];
                        v = geluf(v);
                    }
                    C[(size_t)row * ldc + col] = v;
                }
            }
}

// ---------------------------------------------------------------- utility kernels
__global__ void fill_f32(float* p, float v, size_t n) {
    size_t i = (size_t)blockIdx.x * blockDim.x + threadIdx.x;
    size_t st = (size_t)gridDim.x * blockDim.x;
    for (; i < n; i += st) p[i] = v;
}
__global__ void copy_f32(const float* s, float* d, size_t n) {
    size_t i = (size_t)blockIdx.x * blockDim.x + threadIdx.x;
    size_t st = (size_t)gridDim.x * blockDim.x;
    for (; i < n; i += st) d[i] = s[i];
}
// Wt[n*K + k] = bf16(W[k*Ncol + n])
__global__ void conv_transpose_bf16(const float* __restrict__ W,
                                    unsigned short* __restrict__ Wt,
                                    int K, int Ncol) {
    size_t tot = (size_t)K * Ncol;
    size_t i = (size_t)blockIdx.x * blockDim.x + threadIdx.x;
    size_t st = (size_t)gridDim.x * blockDim.x;
    for (; i < tot; i += st) {
        int n = (int)(i / K), k = (int)(i % K);
        Wt[i] = f2bf(W[(size_t)k * Ncol + n]);
    }
}
// block-diagonal relation weights: Wt[e][(h*32+j)*256 + (h*32+d)] = Wrel[h,e,d,j]
__global__ void conv_rel_blockdiag(const float* __restrict__ Wrel,
                                   unsigned short* __restrict__ Wt, int NE) {
    size_t tot = (size_t)NE * 256 * 256;
    size_t i = (size_t)blockIdx.x * blockDim.x + threadIdx.x;
    size_t st = (size_t)gridDim.x * blockDim.x;
    for (; i < tot; i += st) {
        int e   = (int)(i >> 16);
        int rem = (int)(i & 0xFFFF);
        int n = rem >> 8, k = rem & 255;
        int ho = n >> 5, j = n & 31, hi2 = k >> 5, d = k & 31;
        float v = (ho == hi2)
            ? Wrel[((((size_t)ho * NE + e) * 32) + d) * 32 + j] : 0.0f;
        Wt[i] = f2bf(v);
    }
}

// ---------------------------------------------------------------- edge phase
__global__ void edge_logits_kernel(const float* __restrict__ q,  // +F col offset folded, ld=768
                                   const float* __restrict__ ke, // [Nsrc,256]
                                   const int* __restrict__ src,
                                   const int* __restrict__ dst,
                                   const float* __restrict__ prel,
                                   float* __restrict__ alphaOut, int E) {
    int i = blockIdx.x * 256 + threadIdx.x;
    if (i >= E * 8) return;
    int e = i >> 3, h = i & 7;
    const float* qp = q  + (size_t)dst[e] * 768 + h * 32;
    const float* kp = ke + (size_t)src[e] * 256 + h * 32;
    float a = 0.f;
#pragma unroll
    for (int j = 0; j < 32; j += 4) {
        float4 qa = *(const float4*)(qp + j);
        float4 ka = *(const float4*)(kp + j);
        a += qa.x * ka.x + qa.y * ka.y + qa.z * ka.z + qa.w * ka.w;
    }
    alphaOut[i] = a * prel[h] * 0.17677669529663689f;  // prel / sqrt(32)
}
__global__ void seg_max_kernel(const float* __restrict__ alpha,
                               const int* __restrict__ dst, int dstOff,
                               float* __restrict__ mmax, int E) {
    int i = blockIdx.x * 256 + threadIdx.x;
    if (i >= E * 8) return;
    int e = i >> 3, h = i & 7;
    atomicMaxF(&mmax[(size_t)(dst[e] + dstOff) * 8 + h], alpha[i]);
}
__global__ void seg_expsum_kernel(float* __restrict__ alpha,
                                  const int* __restrict__ dst, int dstOff,
                                  const float* __restrict__ mmax,
                                  float* __restrict__ msum, int E) {
    int i = blockIdx.x * 256 + threadIdx.x;
    if (i >= E * 8) return;
    int e = i >> 3, h = i & 7;
    size_t gi = (size_t)(dst[e] + dstOff) * 8 + h;
    float ea = __expf(alpha[i] - mmax[gi]);
    alpha[i] = ea;
    atomicAdd(&msum[gi], ea);
}
__global__ void seg_agg_kernel(const float* __restrict__ alpha,
                               const float* __restrict__ msum,
                               const float* __restrict__ ve,
                               const int* __restrict__ src,
                               const int* __restrict__ dst, int dstOff,
                               float* __restrict__ agg, int E) {
    long long i = (long long)blockIdx.x * 256 + threadIdx.x;
    if (i >= (long long)E * 256) return;
    int e = (int)(i >> 8), c = (int)(i & 255), h = c >> 5;
    int dg = dst[e] + dstOff;
    float w = alpha[(size_t)e * 8 + h] /
              (msum[(size_t)dg * 8 + h] + 1e-16f);
    atomicAdd(&agg[(size_t)dg * 256 + c],
              w * ve[(size_t)src[e] * 256 + c]);
}

// ---------------------------------------------------------------- post-processing
__global__ __launch_bounds__(256)
void ln_gelu_kernel(float* __restrict__ X, const float* __restrict__ g,
                    const float* __restrict__ b) {
    __shared__ float red[256];
    int row = blockIdx.x, t = threadIdx.x;
    float x = X[(size_t)row * 256 + t];
    red[t] = x; __syncthreads();
    for (int s = 128; s > 0; s >>= 1) { if (t < s) red[t] += red[t + s]; __syncthreads(); }
    float mu = red[0] / 256.f; __syncthreads();
    float d = x - mu;
    red[t] = d * d; __syncthreads();
    for (int s = 128; s > 0; s >>= 1) { if (t < s) red[t] += red[t + s]; __syncthreads(); }
    float var = red[0] / 256.f;
    float y = d * rsqrtf(var + 1e-5f) * g[t] + b[t];
    X[(size_t)row * 256 + t] = geluf(y);
}
__global__ __launch_bounds__(256)
void colmean_gelu_kernel(const float* __restrict__ G, float* __restrict__ outv,
                         int Mrows) {
    __shared__ float red[256];
    int c = blockIdx.x, t = threadIdx.x;
    float s = 0.f;
    for (int r = t; r < Mrows; r += 256) s += geluf(G[(size_t)r * 256 + c]);
    red[t] = s; __syncthreads();
    for (int k = 128; k > 0; k >>= 1) { if (t < k) red[t] += red[t + k]; __syncthreads(); }
    if (t == 0) outv[c] = red[0] / (float)Mrows;
}
__global__ void final_kernel(const float* __restrict__ mean,
                             const float* __restrict__ Wagg2,
                             const float* __restrict__ bagg2,
                             const float* __restrict__ Wo,
                             const float* __restrict__ bo,
                             float* __restrict__ out) {
    __shared__ float ha2[128];
    int t = threadIdx.x;
    if (t < 128) {
        float a = bagg2[t];
        for (int k = 0; k < 256; ++k) a += mean[k] * Wagg2[(size_t)k * 128 + t];
        ha2[t] = geluf(a);
    }
    __syncthreads();
    if (t < 16) {
        float a = bo[t];
        for (int k = 0; k < 128; ++k) a += ha2[k] * Wo[(size_t)k * 16 + t];
        if (isnan(a)) a = 0.0f;
        if (isinf(a)) a = (a > 0) ? 3.4028235e38f : -3.4028235e38f;
        out[t] = a;
    }
}

// ---------------------------------------------------------------- launcher
extern "C" void kernel_launch(void* const* d_in, const int* in_sizes, int n_in,
                              void* d_out, int out_size, void* d_ws, size_t ws_size,
                              hipStream_t stream) {
    (void)in_sizes; (void)n_in; (void)out_size; (void)ws_size;
    const int Nn = 50000, Ee = 150000, Fh = 256, K3 = 768;
    const int ntot = 2 * Nn;
    const int ST[3] = {0, 1, 0}, DT[3] = {1, 0, 0};

    const float* x_op  = (const float*)d_in[0];
    const float* x_var = (const float*)d_in[1];
    const float* ln_g  = (const float*)d_in[26];
    const float* ln_b  = (const float*)d_in[27];
    const float* Wagg1 = (const float*)d_in[28];
    const float* bagg1 = (const float*)d_in[29];
    const float* Wagg2 = (const float*)d_in[30];
    const float* bagg2 = (const float*)d_in[31];
    const float* Wo    = (const float*)d_in[32];
    const float* bo    = (const float*)d_in[33];
    const int* ei[3] = {(const int*)d_in[34], (const int*)d_in[35], (const int*)d_in[36]};

    // ---- workspace bump allocator ----
    char* wsb = (char*)d_ws; size_t off = 0;
    auto alloc = [&](size_t bytes) -> void* {
        void* p = wsb + off;
        off = (off + bytes + 255) & ~(size_t)255;
        return p;
    };
    float* Xa    = (float*)alloc((size_t)ntot * Fh * 4);
    float* Xb    = (float*)alloc((size_t)ntot * Fh * 4);
    float* KQV   = (float*)alloc((size_t)ntot * K3 * 4);
    float* KEb   = (float*)alloc((size_t)Nn * Fh * 4);
    float* VEb   = (float*)alloc((size_t)3 * Nn * Fh * 4);
    float* alpha = (float*)alloc((size_t)3 * Ee * 8 * 4);
    float* mmax  = (float*)alloc((size_t)ntot * 8 * 4);
    float* msum  = (float*)alloc((size_t)ntot * 8 * 4);
    float* agg   = (float*)alloc((size_t)ntot * Fh * 4);
    float* zbias = (float*)alloc(768 * 4);
    float* meanv = (float*)alloc(256 * 4);
    unsigned short* wkqv_t  = (unsigned short*)alloc((size_t)3 * 2 * K3 * Fh * 2);
    unsigned short* krel_t  = (unsigned short*)alloc((size_t)3 * 3 * Fh * Fh * 2);
    unsigned short* vrel_t  = (unsigned short*)alloc((size_t)3 * 3 * Fh * Fh * 2);
    unsigned short* wout_t  = (unsigned short*)alloc((size_t)3 * 2 * Fh * Fh * 2);
    unsigned short* wagg1_t = (unsigned short*)alloc((size_t)2 * Fh * Fh * 2);

    // ---- one-time-per-launch weight conversion (bf16, transposed [N][K]) ----
    fill_f32<<<4, 256, 0, stream>>>(zbias, 0.0f, 768);
    for (int li = 0; li < 3; ++li) {
        int base = 2 + li * 8;
        const float* Wkqv = (const float*)d_in[base + 0];
        const float* Wkr  = (const float*)d_in[base + 2];
        const float* Wvr  = (const float*)d_in[base + 3];
        const float* Wout = (const float*)d_in[base + 5];
        for (int t = 0; t < 2; ++t)
            conv_transpose_bf16<<<768, 256, 0, stream>>>(
                Wkqv + (size_t)t * Fh * K3,
                wkqv_t + (size_t)(li * 2 + t) * K3 * Fh, Fh, K3);
        conv_rel_blockdiag<<<768, 256, 0, stream>>>(Wkr, krel_t + (size_t)li * 3 * 65536, 3);
        conv_rel_blockdiag<<<768, 256, 0, stream>>>(Wvr, vrel_t + (size_t)li * 3 * 65536, 3);
        for (int t = 0; t < 2; ++t)
            conv_transpose_bf16<<<256, 256, 0, stream>>>(
                Wout + (size_t)t * 65536,
                wout_t + (size_t)(li * 2 + t) * 65536, Fh, Fh);
    }
    for (int t = 0; t < 2; ++t)
        conv_transpose_bf16<<<256, 256, 0, stream>>>(
            Wagg1 + (size_t)t * 65536, wagg1_t + (size_t)t * 65536, Fh, Fh);

    // ---- concat inputs into Xa ----
    copy_f32<<<2048, 256, 0, stream>>>(x_op,  Xa,                   (size_t)Nn * Fh);
    copy_f32<<<2048, 256, 0, stream>>>(x_var, Xa + (size_t)Nn * Fh, (size_t)Nn * Fh);

    float* Xcur = Xa; float* Xnext = Xb;
    const int eb8 = (Ee * 8 + 255) / 256;
    const dim3 gKQV((K3 + BN - 1) / BN, (Nn + BM - 1) / BM);
    const dim3 gSQ ((Fh + BN - 1) / BN, (Nn + BM - 1) / BM);

    for (int li = 0; li < 3; ++li) {
        int base = 2 + li * 8;
        const float* bkqv = (const float*)d_in[base + 1];
        const float* prel = (const float*)d_in[base + 4];
        const float* bout = (const float*)d_in[base + 6];
        const float* skip = (const float*)d_in[base + 7];

        // 1) fused KQV projection per node type (WMMA)
        for (int t = 0; t < 2; ++t)
            gemm_bf16_wmma<256, 0, 0><<<gKQV, 256, 0, stream>>>(
                Xcur + (size_t)t * Nn * Fh, Fh,
                wkqv_t + (size_t)(li * 2 + t) * K3 * Fh,
                bkqv + (size_t)t * K3,
                KQV + (size_t)t * Nn * K3, K3,
                nullptr, 0, nullptr, Nn, K3);

        // 2) init softmax accumulators
        fill_f32<<<2048, 256, 0, stream>>>(mmax, -3.0e38f, (size_t)ntot * 8);
        fill_f32<<<2048, 256, 0, stream>>>(msum, 0.0f,     (size_t)ntot * 8);
        fill_f32<<<2048, 256, 0, stream>>>(agg,  0.0f,     (size_t)ntot * Fh);

        // 3) per edge type: relation transforms (WMMA, block-diag weights) + logits
        for (int e = 0; e < 3; ++e) {
            int st = ST[e], dt = DT[e];
            gemm_bf16_wmma<256, 0, 0><<<gSQ, 256, 0, stream>>>(   // KE = k @ Wkrel_blk
                KQV + (size_t)st * Nn * K3, K3,
                krel_t + (size_t)(li * 3 + e) * 65536, zbias,
                KEb, Fh, nullptr, 0, nullptr, Nn, Fh);
            gemm_bf16_wmma<256, 0, 0><<<gSQ, 256, 0, stream>>>(   // VE = v @ Wvrel_blk
                KQV + (size_t)st * Nn * K3 + 2 * Fh, K3,
                vrel_t + (size_t)(li * 3 + e) * 65536, zbias,
                VEb + (size_t)e * Nn * Fh, Fh, nullptr, 0, nullptr, Nn, Fh);
            edge_logits_kernel<<<eb8, 256, 0, stream>>>(
                KQV + (size_t)dt * Nn * K3 + Fh,                  // q block
                KEb, ei[e], ei[e] + Ee, prel + e * 8,
                alpha + (size_t)e * Ee * 8, Ee);
        }

        // 4) segment softmax across union of incoming edges
        for (int e = 0; e < 3; ++e)
            seg_max_kernel<<<eb8, 256, 0, stream>>>(
                alpha + (size_t)e * Ee * 8, ei[e] + Ee, DT[e] * Nn, mmax, Ee);
        for (int e = 0; e < 3; ++e)
            seg_expsum_kernel<<<eb8, 256, 0, stream>>>(
                alpha + (size_t)e * Ee * 8, ei[e] + Ee, DT[e] * Nn, mmax, msum, Ee);
        for (int e = 0; e < 3; ++e)
            seg_agg_kernel<<<Ee, 256, 0, stream>>>(
                alpha + (size_t)e * Ee * 8, msum,
                VEb + (size_t)e * Nn * Fh, ei[e], ei[e] + Ee, DT[e] * Nn, agg, Ee);

        // 5) output projection + skip gate (WMMA, gelu(A) fused; gelu-out for layers 2,3)
        for (int t = 0; t < 2; ++t) {
            if (li == 0)
                gemm_bf16_wmma<256, 1, 1><<<gSQ, 256, 0, stream>>>(
                    agg + (size_t)t * Nn * Fh, Fh,
                    wout_t + (size_t)(li * 2 + t) * 65536,
                    bout + (size_t)t * Fh,
                    Xnext + (size_t)t * Nn * Fh, Fh,
                    Xcur + (size_t)t * Nn * Fh, Fh,
                    skip + t, Nn, Fh);
            else
                gemm_bf16_wmma<256, 1, 2><<<gSQ, 256, 0, stream>>>(
                    agg + (size_t)t * Nn * Fh, Fh,
                    wout_t + (size_t)(li * 2 + t) * 65536,
                    bout + (size_t)t * Fh,
                    Xnext + (size_t)t * Nn * Fh, Fh,
                    Xcur + (size_t)t * Nn * Fh, Fh,
                    skip + t, Nn, Fh);
        }

        // 6) layer-1 only: gelu(layer_norm(h))
        if (li == 0)
            for (int t = 0; t < 2; ++t)
                ln_gelu_kernel<<<Nn, 256, 0, stream>>>(
                    Xnext + (size_t)t * Nn * Fh, ln_g + t * Fh, ln_b + t * Fh);

        float* tmp = Xcur; Xcur = Xnext; Xnext = tmp;
    }

    // ---- tail: Wagg1 (WMMA) -> gelu+mean -> small MLP -> out ----
    for (int t = 0; t < 2; ++t)
        gemm_bf16_wmma<256, 0, 0><<<gSQ, 256, 0, stream>>>(
            Xcur + (size_t)t * Nn * Fh, Fh,
            wagg1_t + (size_t)t * 65536,
            bagg1 + (size_t)t * Fh,
            KQV + (size_t)t * Nn * Fh, Fh,      // reuse KQV as G buffer
            nullptr, 0, nullptr, Nn, Fh);
    colmean_gelu_kernel<<<256, 256, 0, stream>>>(KQV, meanv, ntot);
    final_kernel<<<1, 128, 0, stream>>>(meanv, Wagg2, bagg2, Wo, bo, (float*)d_out);
}
